// SSIMLoss_39367670235406
// MI455X (gfx1250) — matrix operand order repaired and checked by reading, not compiled, not run on hardware
//
#include <hip/hip_runtime.h>
#include <hip/hip_bf16.h>

typedef _Float16 h16;
typedef __attribute__((ext_vector_type(16))) _Float16 v16h;
typedef __attribute__((ext_vector_type(8)))  _Float16 v8h;
typedef __attribute__((ext_vector_type(8)))  float    v8f;

#define IMG_H 1024
#define IMG_W 1024
#define TR 64            // output tile rows per workgroup
#define TC 64            // output tile cols per workgroup
#define RAD 5
#define LI 80            // loaded rows  (TR + 2*RAD = 74, padded to 80)
#define LJ 80            // loaded cols
#define FSTR 88          // stride (halves) over li in F  (col-major [lj][li]); 88*2B = 176B, 16B-aligned rows
#define VSTR 88          // stride (halves) over c in VF  (row-major [r][c])
#define C1F 4.0e-4f      // (0.01*2)^2
#define C2F 3.6e-3f      // (0.03*2)^2

__device__ __forceinline__ v8f wmma16(v16h a, v16h b, v8f c) {
  // D = A(16x32 f16) x B(32x16 f16) + C(16x16 f32)
  return __builtin_amdgcn_wmma_f32_16x16x32_f16(false, a, false, b, (short)0, c, false, false);
}

// A-operand layout (16-bit A 16x32, wave32):
//  lanes 0-15 : halves[0..7] = K 0..7,  halves[8..15] = K 16..23   (M = lane)
//  lanes 16-31: halves[0..7] = K 8..15, halves[8..15] = K 24..31   (M = lane-16)
// base points at K=0 of this lane's row; runs are 16B aligned.
__device__ __forceinline__ v16h load_a(const h16* base, int sel) {
  const v8h lo = *(const v8h*)(base + 8 * sel);
  const v8h hi = *(const v8h*)(base + 16 + 8 * sel);
  v16h r;
#pragma unroll
  for (int i = 0; i < 8; ++i) { r[i] = lo[i]; r[8 + i] = hi[i]; }
  return r;
}

__device__ __forceinline__ v8h cvt8(v8f d) {
  v8h o;
#pragma unroll
  for (int i = 0; i < 8; ++i) o[i] = (h16)d[i];
  return o;
}

__global__ __launch_bounds__(256)
void ssim_tiles(const float* __restrict__ img1, const float* __restrict__ img2,
                float* __restrict__ partial) {
  __shared__ __attribute__((aligned(16))) h16 F1[LJ * FSTR];      // img1 tile, [col][row]
  __shared__ __attribute__((aligned(16))) h16 F2[LJ * FSTR];      // img2 tile, [col][row]
  __shared__ __attribute__((aligned(16))) h16 VF[5][TR * VSTR];   // vertically blurred fields, [row][col]
  __shared__ float red[256];

  const int tid  = threadIdx.x;
  const int wave = tid >> 5;   // wave32
  const int lane = tid & 31;
  const int nn   = lane & 15;
  const int sel  = lane >> 4;

  // Constant banded-Gaussian B operand, identical for both passes: B[K][N] = g[K-N]
  // B layout (16-bit 32x16, wave32): lane holds column N=lane&15; halves j=0..15 are
  // rows K = 8*sel + (j&7) + 16*(j>>3).
  v16h bw;
#pragma unroll
  for (int j = 0; j < 16; ++j) {
    const int t = (j & 7) + 8 * sel + 16 * (j >> 3) - nn;
    float w = 0.0f;
    if (t >= 0 && t <= 10) {
      const float d = (float)(t - 5);
      w = expf(-d * d * (1.0f / 4.5f)) * 0.26601171f;  // normalized gaussian, sigma=1.5
    }
    bw[j] = (h16)w;
  }

  const int r0 = (int)blockIdx.y * TR - RAD;
  const int c0 = (int)blockIdx.x * TC - RAD;
  const float* p1 = img1 + (size_t)blockIdx.z * (IMG_H * IMG_W);
  const float* p2 = img2 + (size_t)blockIdx.z * (IMG_H * IMG_W);

  // ---- load (zero-padded at image borders), f32 -> f16, transposed into LDS ----
  for (int idx = tid; idx < LI * LJ; idx += 256) {
    const int li = idx / LJ;        // image-row offset in tile
    const int lj = idx - li * LJ;   // image-col offset in tile (contiguous per tid -> coalesced)
    const int gi = r0 + li;
    const int gj = c0 + lj;
    const bool ok = (gi >= 0) & (gi < IMG_H) & (gj >= 0) & (gj < IMG_W);
    const float x1 = ok ? p1[gi * IMG_W + gj] : 0.0f;
    const float x2 = ok ? p2[gi * IMG_W + gj] : 0.0f;
    F1[lj * FSTR + li] = (h16)x1;
    F2[lj * FSTR + li] = (h16)x2;
  }
  __syncthreads();

  // ---- vertical pass: voutT[16cols x 16rows] = (F^T)[16x32] x G[32x16] ----
  // 5 col-chunks (80 cols) x 4 row-chunks (64 out rows); squares built in registers.
  for (int t = wave; t < 20; t += 8) {
    const int cc = t >> 2;           // 0..4
    const int rc = t & 3;            // 0..3
    const int c  = cc * 16 + nn;     // A-matrix row M = tile column
    const v16h a1 = load_a(&F1[c * FSTR + rc * 16], sel);
    const v16h a2 = load_a(&F2[c * FSTR + rc * 16], sel);
    const v16h a11 = a1 * a1;        // v_pk_mul_f16
    const v16h a22 = a2 * a2;
    const v16h a12 = a1 * a2;
    const v8f z = {};
    const v8f d0 = wmma16(a1,  bw, z);
    const v8f d1 = wmma16(a2,  bw, z);
    const v8f d2 = wmma16(a11, bw, z);
    const v8f d3 = wmma16(a22, bw, z);
    const v8f d4 = wmma16(a12, bw, z);
    // D layout: lane holds column N=nn (= output row r), VGPR v holds M=v+8*sel (= tile col).
    const int r     = rc * 16 + nn;
    const int cbase = cc * 16 + 8 * sel;
    *(v8h*)(&VF[0][r * VSTR + cbase]) = cvt8(d0);
    *(v8h*)(&VF[1][r * VSTR + cbase]) = cvt8(d1);
    *(v8h*)(&VF[2][r * VSTR + cbase]) = cvt8(d2);
    *(v8h*)(&VF[3][r * VSTR + cbase]) = cvt8(d3);
    *(v8h*)(&VF[4][r * VSTR + cbase]) = cvt8(d4);
  }
  __syncthreads();

  // ---- horizontal pass + elementwise SSIM ----
  float acc = 0.0f;
  for (int t = wave; t < 16; t += 8) {
    const int hc  = t & 3;
    const int hr  = t >> 2;
    const int row = hr * 16 + nn;    // A-matrix row M = output row
    const int ab  = row * VSTR + hc * 16;
    const v8f z = {};
    const v8f mu1 = wmma16(load_a(&VF[0][ab], sel), bw, z);
    const v8f mu2 = wmma16(load_a(&VF[1][ab], sel), bw, z);
    const v8f exx = wmma16(load_a(&VF[2][ab], sel), bw, z);
    const v8f eyy = wmma16(load_a(&VF[3][ab], sel), bw, z);
    const v8f exy = wmma16(load_a(&VF[4][ab], sel), bw, z);
#pragma unroll
    for (int i = 0; i < 8; ++i) {
      const float m1 = mu1[i], m2 = mu2[i];
      const float m11 = m1 * m1, m22 = m2 * m2, m12 = m1 * m2;
      const float s11 = exx[i] - m11;
      const float s22 = eyy[i] - m22;
      const float s12 = exy[i] - m12;
      const float num = (2.0f * m12 + C1F) * (2.0f * s12 + C2F);
      const float den = (m11 + m22 + C1F) * (s11 + s22 + C2F);
      acc += num / den;
    }
  }

  // ---- deterministic in-block tree reduction ----
  red[tid] = acc;
  __syncthreads();
  for (int s = 128; s > 0; s >>= 1) {
    if (tid < s) red[tid] += red[tid + s];
    __syncthreads();
  }
  if (tid == 0) {
    const int wg = ((int)blockIdx.z * gridDim.y + blockIdx.y) * gridDim.x + blockIdx.x;
    partial[wg] = red[0];
  }
}

__global__ __launch_bounds__(256)
void ssim_reduce(const float* __restrict__ partial, float* __restrict__ out,
                 int n, float inv_pix) {
  __shared__ float red[256];
  float s = 0.0f;
  for (int i = threadIdx.x; i < n; i += 256) s += partial[i];   // fixed order -> deterministic
  red[threadIdx.x] = s;
  __syncthreads();
  for (int k = 128; k > 0; k >>= 1) {
    if ((int)threadIdx.x < k) red[threadIdx.x] += red[threadIdx.x + k];
    __syncthreads();
  }
  if (threadIdx.x == 0) out[0] = 1.0f - red[0] * inv_pix;
}

extern "C" void kernel_launch(void* const* d_in, const int* in_sizes, int n_in,
                              void* d_out, int out_size, void* d_ws, size_t ws_size,
                              hipStream_t stream) {
  (void)n_in; (void)out_size; (void)ws_size;
  const float* img1 = (const float*)d_in[0];
  const float* img2 = (const float*)d_in[1];
  float* ws  = (float*)d_ws;
  float* out = (float*)d_out;

  const int batches = in_sizes[0] / (IMG_H * IMG_W);   // 32
  dim3 grid(IMG_W / TC, IMG_H / TR, batches);          // 16 x 16 x 32 = 8192 WGs
  const int nwg = (int)(grid.x * grid.y * grid.z);

  ssim_tiles<<<grid, 256, 0, stream>>>(img1, img2, ws);
  ssim_reduce<<<1, 256, 0, stream>>>(ws, out, nwg, 1.0f / (float)in_sizes[0]);
}
